// ScaledDotProductAttention_17995912970900
// MI455X (gfx1250) — matrix operand order repaired and checked by reading, not compiled, not run on hardware
//
#include <hip/hip_runtime.h>
#include <cstdint>
#include <cstddef>

// Problem constants (match reference): B=4, H=16, S=2048, DK=64, fp32.
#define B_  4
#define H_  16
#define S_  2048
#define DK_ 64

#define BM     32                 // query rows per workgroup
#define CHUNK  128                // keys/values staged per LDS chunk
#define NCHUNK (S_ / CHUNK)       // 16
#define SS_STRIDE (S_ + 4)        // f32 slab stride: 2052 (stride%64banks=4 -> conflict-free)
#define QH_STRIDE (DK_ + 8)       // 72 halfs (144B rows, 16B aligned, bank-spread)
#define KH_STRIDE (DK_ + 8)       // 72 halfs
#define VT_STRIDE (CHUNK + 8)     // 136 halfs (272B rows)

typedef __attribute__((ext_vector_type(8)))  float    v8f;
typedef __attribute__((ext_vector_type(8)))  _Float16 v8h;
typedef __attribute__((ext_vector_type(16))) _Float16 v16h;

// dynamic LDS layout:
//   Ss   f32 [BM * SS_STRIDE]            scores -> probabilities (full precision)
//   Qh   f16 [BM * QH_STRIDE]            Q tile
//   KVh  f16 [max(CHUNK*KH_STRIDE, DK_*VT_STRIDE)]   staged K (row-major) or V^T
//   red  f32 [256], rinv f32 [BM]
#define KVH_HALFS ((CHUNK*KH_STRIDE) > (DK_*VT_STRIDE) ? (CHUNK*KH_STRIDE) : (DK_*VT_STRIDE))
#define SMEM_BYTES ( BM*SS_STRIDE*4 + BM*QH_STRIDE*2 + KVH_HALFS*2 + 256*4 + BM*4 )

__device__ __forceinline__ v16h ld_op16(const _Float16* p_lo, const _Float16* p_hi) {
    v8h lo = *(const v8h*)p_lo;     // ds_load_b128
    v8h hi = *(const v8h*)p_hi;     // ds_load_b128
    return __builtin_shufflevector(lo, hi, 0,1,2,3,4,5,6,7,8,9,10,11,12,13,14,15);
}

__global__ __launch_bounds__(256)
void attn_fused_f16wmma(const float* __restrict__ Q,
                        const float* __restrict__ K,
                        const float* __restrict__ V,
                        const unsigned char* __restrict__ mask,  // jnp.bool_ = 1 byte
                        float* __restrict__ out,
                        float* __restrict__ attn)
{
    extern __shared__ float smem[];
    float*    Ss   = smem;                                   // 262,656 B
    _Float16* Qh   = (_Float16*)(Ss + BM * SS_STRIDE);       // 4,608 B
    _Float16* KVh  = Qh + BM * QH_STRIDE;                    // 18,432 B
    float*    red  = (float*)(KVh + KVH_HALFS);
    float*    rinv = red + 256;

    const int tid   = threadIdx.x;
    const int lane  = tid & 31;
    const int wave  = tid >> 5;       // 0..7
    const int lhalf = lane >> 4;      // 0 | 1
    const int l16   = lane & 15;

    const int nQB = S_ / BM;                  // 64
    const int qb  = (int)blockIdx.x % nQB;
    const int bh  = (int)blockIdx.x / nQB;
    const int b   = bh / H_;
    const int q0  = qb * BM;

    const float* Qg = Q + ((size_t)bh * S_ + q0) * DK_;
    const float* Kg = K + (size_t)bh * S_ * DK_;
    const float* Vg = V + (size_t)bh * S_ * DK_;
    const unsigned char* Mg = mask + ((size_t)b * S_ + q0) * S_;
    float* Og = out  + ((size_t)bh * S_ + q0) * DK_;
    float* Ag = attn + ((size_t)bh * S_ + q0) * S_;

    // ---------------- load Q tile (32x64) -> f16 LDS ----------------
    for (int i = tid; i < BM * (DK_ / 4); i += 256) {
        const int r = i >> 4, c = (i & 15) << 2;
        const float4 v = *(const float4*)(Qg + (size_t)r * DK_ + c);
        _Float16* d = Qh + r * QH_STRIDE + c;
        d[0] = (_Float16)v.x; d[1] = (_Float16)v.y;
        d[2] = (_Float16)v.z; d[3] = (_Float16)v.w;     // merges to ds_store_b64
    }

    const float scale = 0.125f;  // 1/sqrt(64)

    // ---------------- scores: Ss = mask(scale * Q K^T), f16 WMMA / f32 acc ----------------
    for (int ch = 0; ch < NCHUNK; ++ch) {
        __syncthreads();
        // stage K chunk [128 x 64] f32 -> f16 row-major (B(k,n)=K[n][k] is contiguous in k)
        for (int i = tid; i < CHUNK * (DK_ / 4); i += 256) {
            const int r = i >> 4, c = (i & 15) << 2;
            const float4 v = *(const float4*)(Kg + ((size_t)(ch * CHUNK + r)) * DK_ + c);
            _Float16* d = KVh + r * KH_STRIDE + c;
            d[0] = (_Float16)v.x; d[1] = (_Float16)v.y;
            d[2] = (_Float16)v.z; d[3] = (_Float16)v.w;
        }
        if (ch + 1 < NCHUNK)
            __builtin_prefetch(Kg + ((size_t)(ch + 1) * CHUNK + (tid >> 1)) * DK_, 0, 1);
        __syncthreads();

        const int nloc = wave * 16;               // this wave's 16-key column tile
        const int n    = nloc + l16;
        for (int mt = 0; mt < 2; ++mt) {
            const int m = mt * 16 + l16;
            v8f acc = {};
            #pragma unroll
            for (int ks = 0; ks < DK_; ks += 32) {
                // A[16x32] f16: elems 0-7 <- K=ks+8h.., elems 8-15 <- K=ks+16+8h..
                const v16h a = ld_op16(Qh + m * QH_STRIDE + ks + 8 * lhalf,
                                       Qh + m * QH_STRIDE + ks + 16 + 8 * lhalf);
                // B[32x16] f16: elems 0-7 <- K[n][ks+8h..], elems 8-15 <- K[n][ks+16+8h..]
                const v16h bm = ld_op16(KVh + n * KH_STRIDE + ks + 8 * lhalf,
                                        KVh + n * KH_STRIDE + ks + 16 + 8 * lhalf);
                acc = __builtin_amdgcn_wmma_f32_16x16x32_f16(
                          false, a, false, bm, (short)0, acc, false, false);
            }
            // D layout: VGPR r -> M = mt*16 + r + 8*lhalf, N = l16
            const int ncol = ch * CHUNK + nloc + l16;
            #pragma unroll
            for (int r = 0; r < 8; ++r) {
                const int mrow = mt * 16 + r + lhalf * 8;
                float v = acc[r] * scale;
                if (Mg[(size_t)mrow * S_ + ncol]) v = -1e9f;
                Ss[mrow * SS_STRIDE + ncol] = v;
            }
        }
    }
    __syncthreads();

    // ---------------- row softmax over Ss [32 x 2048], all f32 ----------------
    {
        const int row = tid >> 3;          // 8 threads per row
        const int seg = tid & 7;
        const int c0  = seg * (S_ / 8);
        float m = -3.402823466e+38f;
        for (int c = 0; c < S_ / 8; ++c)
            m = fmaxf(m, Ss[row * SS_STRIDE + c0 + c]);
        red[tid] = m;
        __syncthreads();
        float rm = red[row * 8];
        #pragma unroll
        for (int t = 1; t < 8; ++t) rm = fmaxf(rm, red[row * 8 + t]);
        __syncthreads();
        float s = 0.f;
        for (int c = 0; c < S_ / 8; ++c) {
            const float e = __expf(Ss[row * SS_STRIDE + c0 + c] - rm);
            Ss[row * SS_STRIDE + c0 + c] = e;
            s += e;
        }
        red[tid] = s;
        __syncthreads();
        if (seg == 0) {
            float t = 0.f;
            #pragma unroll
            for (int u = 0; u < 8; ++u) t += red[row * 8 + u];
            rinv[row] = 1.0f / t;
        }
        __syncthreads();
    }

    // ---------------- normalize in LDS + stream attn (f32) to HBM ----------------
    for (int i = tid; i < BM * S_ / 4; i += 256) {
        const int e   = i << 2;
        const int row = e >> 11;
        const int col = e & (S_ - 1);
        const float inv = rinv[row];
        float4 p = *(const float4*)(Ss + row * SS_STRIDE + col);
        p.x *= inv; p.y *= inv; p.z *= inv; p.w *= inv;
        *(float4*)(Ss + row * SS_STRIDE + col) = p;
        *(float4*)(Ag + (size_t)row * S_ + col) = p;
    }

    // ---------------- out = P @ V : A = f32 P (cvt on the fly), B = f16 V^T ----------------
    _Float16* Vt = KVh;                  // [DK_][VT_STRIDE] : Vt[n][k] = V[k][n]
    const int mt = wave >> 2;            // 2 row tiles
    const int n0 = (wave & 3) * 16;      // 4 col tiles of DK
    const int nn = n0 + l16;
    v8f oacc = {};
    for (int ch = 0; ch < NCHUNK; ++ch) {
        __syncthreads();
        // stage V chunk transposed: coalesced global float4 reads, f16 scattered LDS stores
        for (int i = tid; i < CHUNK * (DK_ / 4); i += 256) {
            const int r = i >> 4;            // local k row
            const int c = (i & 15) << 2;     // dk col
            const float4 v = *(const float4*)(Vg + ((size_t)(ch * CHUNK + r)) * DK_ + c);
            Vt[(c + 0) * VT_STRIDE + r] = (_Float16)v.x;
            Vt[(c + 1) * VT_STRIDE + r] = (_Float16)v.y;
            Vt[(c + 2) * VT_STRIDE + r] = (_Float16)v.z;
            Vt[(c + 3) * VT_STRIDE + r] = (_Float16)v.w;
        }
        if (ch + 1 < NCHUNK)
            __builtin_prefetch(Vg + ((size_t)(ch + 1) * CHUNK + (tid >> 1)) * DK_, 0, 1);
        __syncthreads();

        const int m = mt * 16 + l16;
        const float* ap = Ss + m * SS_STRIDE + ch * CHUNK;
        #pragma unroll
        for (int kk = 0; kk < CHUNK / 32; ++kk) {
            const int kc = kk * 32;
            // A: convert f32 probabilities -> f16 in registers (v_cvt co-executes with WMMA)
            v16h a;
            #pragma unroll
            for (int j = 0; j < 8; ++j)
                a[j] = (_Float16)ap[kc + 8 * lhalf + j];          // ds_load_b128 x2
            #pragma unroll
            for (int j = 0; j < 8; ++j)
                a[8 + j] = (_Float16)ap[kc + 16 + 8 * lhalf + j]; // ds_load_b128 x2
            // B: elems 0-7 <- Vt[n][kc+8h..], elems 8-15 <- Vt[n][kc+16+8h..]
            const v16h bm = ld_op16(Vt + nn * VT_STRIDE + kc + 8 * lhalf,
                                    Vt + nn * VT_STRIDE + kc + 16 + 8 * lhalf);
            oacc = __builtin_amdgcn_wmma_f32_16x16x32_f16(
                       false, a, false, bm, (short)0, oacc, false, false);
        }
    }
    #pragma unroll
    for (int r = 0; r < 8; ++r) {
        const int mrow = mt * 16 + r + lhalf * 8;
        Og[(size_t)mrow * DK_ + n0 + l16] = oacc[r];
    }
}

extern "C" void kernel_launch(void* const* d_in, const int* in_sizes, int n_in,
                              void* d_out, int out_size, void* d_ws, size_t ws_size,
                              hipStream_t stream) {
    (void)in_sizes; (void)n_in; (void)out_size; (void)d_ws; (void)ws_size;
    const float* Q = (const float*)d_in[0];
    const float* K = (const float*)d_in[1];
    const float* V = (const float*)d_in[2];
    const unsigned char* mask = (const unsigned char*)d_in[3];

    float* out  = (float*)d_out;                          // [B,H,S,DK]
    float* attn = out + (size_t)B_ * H_ * S_ * DK_;       // [B,H,S,S]

    static_assert(SMEM_BYTES <= 320u * 1024u, "LDS budget");
    const size_t shmem = SMEM_BYTES;                      // ~287 KB

    (void)hipFuncSetAttribute((const void*)attn_fused_f16wmma,
                              hipFuncAttributeMaxDynamicSharedMemorySize,
                              (int)shmem);

    const dim3 grid(B_ * H_ * (S_ / BM));                 // 4096 workgroups
    attn_fused_f16wmma<<<grid, 256, shmem, stream>>>(Q, K, V, mask, out, attn);
}
